// CPUMaxEfficiencyMoE_31920196944053
// MI455X (gfx1250) — matrix operand order repaired and checked by reading, not compiled, not run on hardware
//
#include <hip/hip_runtime.h>

// ---------------------------------------------------------------------------
// MoE (top-2 of 8 experts, d_model = d_ff = 1024, N = 8192 tokens)
// Routed implementation (identical math: non-selected experts have comb==0).
// v_wmma_f32_16x16x32_f16 GEMMs; weight panels cooperatively converted to f16
// and staged transposed in LDS; all fragments load as ds_load_b128, with
// B-fragment loads software-pipelined 2 tiles ahead of the WMMAs.
// ---------------------------------------------------------------------------

#define NEXP   8
#define DMODEL 1024
#define DFF    1024
#define NTOK   8192            // B*T = 4*2048
#define BM     32              // tokens per block (2 row-groups of 16)
#define NBLK   (NTOK / BM)     // 256 worst-case row-blocks per expert
#define BPAD   40              // s_B row stride in f16: 80 B, 16B-aligned,
                               // col*20 dwords mod 64 distinct -> conflict-free reads
#define MPAD   (DFF + 8)       // s_mid row stride: 1032 f16 = 2064 B (16B multiple)

typedef _Float16 v16h __attribute__((ext_vector_type(16)));
typedef _Float16 v8h  __attribute__((ext_vector_type(8)));
typedef float    v8f  __attribute__((ext_vector_type(8)));

static __device__ __forceinline__ v16h cat8(v8h a, v8h b) {
  return __builtin_shufflevector(a, b, 0,1,2,3,4,5,6,7,8,9,10,11,12,13,14,15);
}
// B fragment: 16 contiguous f16 (K = half*16 + 0..15) -> two ds_load_b128
static __device__ __forceinline__ v16h ldfragB(const _Float16* p) {
  const v8h* q = (const v8h*)p;
  return cat8(q[0], q[1]);
}
// A fragment: 16 contiguous f16 (K = half*16 + 0..15)
static __device__ __forceinline__ v16h ldfragA(const _Float16* p) {
  const v8h* q = (const v8h*)p;
  return cat8(q[0], q[1]);
}

// ---------------------------------------------------------------------------
// Kernel 1: zero output (harness poisons d_out) and per-expert counters.
// 4096 blocks * 256 threads * 8 floats = 8,388,608 = NTOK*DMODEL exactly.
// ---------------------------------------------------------------------------
__global__ __launch_bounds__(256) void moe_zero_kernel(float* __restrict__ out,
                                                       int* __restrict__ counts) {
  const size_t tid  = (size_t)blockIdx.x * blockDim.x + threadIdx.x;
  const size_t base = tid * 8;
  float4 z = make_float4(0.f, 0.f, 0.f, 0.f);
  *(float4*)(out + base)     = z;
  *(float4*)(out + base + 4) = z;
  if (blockIdx.x == 0 && threadIdx.x < NEXP) counts[threadIdx.x] = 0;
}

// ---------------------------------------------------------------------------
// Kernel 2: router. One wave32 per token: logits = x[n] @ Wr, softmax, top-2,
// atomic-append (token, weight) into per-expert gather lists.
// ---------------------------------------------------------------------------
__global__ __launch_bounds__(256) void moe_router_kernel(
    const float* __restrict__ x, const float* __restrict__ Wr,
    int* __restrict__ counts, int* __restrict__ lists, float* __restrict__ wts) {
  const int tid  = threadIdx.x;
  const int wave = tid >> 5;
  const int lane = tid & 31;
  const int n    = blockIdx.x * 8 + wave;   // 8 waves per block

  float a[NEXP];
  #pragma unroll
  for (int e = 0; e < NEXP; ++e) a[e] = 0.f;

  const float* xrow = x + (size_t)n * DMODEL;
  for (int d = lane; d < DMODEL; d += 32) {
    const float  xv = xrow[d];
    const float4 w0 = *(const float4*)(Wr + (size_t)d * NEXP);
    const float4 w1 = *(const float4*)(Wr + (size_t)d * NEXP + 4);
    a[0] += xv * w0.x; a[1] += xv * w0.y; a[2] += xv * w0.z; a[3] += xv * w0.w;
    a[4] += xv * w1.x; a[5] += xv * w1.y; a[6] += xv * w1.z; a[7] += xv * w1.w;
  }

  #pragma unroll
  for (int off = 16; off > 0; off >>= 1) {
    #pragma unroll
    for (int e = 0; e < NEXP; ++e) a[e] += __shfl_xor(a[e], off, 32);
  }

  if (lane == 0) {
    float mx = a[0];
    #pragma unroll
    for (int e = 1; e < NEXP; ++e) mx = fmaxf(mx, a[e]);
    float p[NEXP], s = 0.f;
    #pragma unroll
    for (int e = 0; e < NEXP; ++e) { p[e] = __expf(a[e] - mx); s += p[e]; }
    int i0 = 0;                               // lowest index on ties, like lax.top_k
    #pragma unroll
    for (int e = 1; e < NEXP; ++e) if (p[e] > p[i0]) i0 = e;
    int i1 = (i0 == 0) ? 1 : 0;
    #pragma unroll
    for (int e = 0; e < NEXP; ++e) if (e != i0 && p[e] > p[i1]) i1 = e;
    const float inv = 1.f / s;
    const float w0 = p[i0] * inv, w1 = p[i1] * inv;
    int s0 = atomicAdd(&counts[i0], 1);
    lists[i0 * NTOK + s0] = n; wts[i0 * NTOK + s0] = w0;
    int s1 = atomicAdd(&counts[i1], 1);
    lists[i1 * NTOK + s1] = n; wts[i1 * NTOK + s1] = w1;
  }
}

// ---------------------------------------------------------------------------
// Cooperative weight-panel stage: src = 32 rows x 1024 cols (row-major fp32),
// converted to f16 and stored TRANSPOSED: sB[col][k].  Coalesced b128 global
// loads; k-adjacent pairs packed and written as single b32 LDS stores.
// ---------------------------------------------------------------------------
static __device__ __forceinline__ void stage_B(const float* __restrict__ src,
                                               _Float16 (*sB)[BPAD], int tid) {
  #pragma unroll
  for (int rp = 0; rp < 16; ++rp) {           // k-row pair (2*rp, 2*rp+1)
    const float4 g0 = *(const float4*)(src + (size_t)(2 * rp)     * 1024 + tid * 4);
    const float4 g1 = *(const float4*)(src + (size_t)(2 * rp + 1) * 1024 + tid * 4);
    const float a0[4] = {g0.x, g0.y, g0.z, g0.w};
    const float a1[4] = {g1.x, g1.y, g1.z, g1.w};
    #pragma unroll
    for (int i = 0; i < 4; ++i) {
      union { _Float16 h[2]; unsigned u; } p;
      p.h[0] = (_Float16)a0[i];               // k = 2*rp
      p.h[1] = (_Float16)a1[i];               // k = 2*rp+1
      ((unsigned*)&sB[tid * 4 + i][0])[rp] = p.u;
    }
  }
}

// ---------------------------------------------------------------------------
// Kernel 3: fused routed expert MLP.
// Block = (expert e, 32 gathered tokens), 256 threads = 8 waves arranged as
// 2 row-groups x 4 col-groups; wave tile = 16 rows x 256 cols (16 WMMA tiles).
//   stage x_gather -> s_mid (f16)
//   GEMM1: mid = sqrelu(x @ W1[e])   (A from LDS, B panel staged per k-step)
//   GEMM2: y   = mid @ W2[e];  out[tok] += weight * y  (atomic f32)
// ---------------------------------------------------------------------------
__global__ __launch_bounds__(256) void moe_expert_kernel(
    const float* __restrict__ x,  const float* __restrict__ W1,
    const float* __restrict__ W2, const int* __restrict__ counts,
    const int* __restrict__ lists, const float* __restrict__ wts,
    float* __restrict__ out) {
  __shared__ __attribute__((aligned(16))) _Float16 s_B[1024][BPAD];   // 80 KB
  __shared__ __attribute__((aligned(16))) _Float16 s_mid[BM][MPAD];   // 66 KB
  __shared__ int      s_tok[BM];
  __shared__ float    s_w[BM];

  const int e    = blockIdx.y;
  const int cnt  = counts[e];
  const int row0 = blockIdx.x * BM;
  if (row0 >= cnt) return;                    // uniform: whole block exits

  const int tid = threadIdx.x;
  if (tid < BM) {
    int tok = 0; float w = 0.f;               // padding rows: token 0, weight 0
    if (row0 + tid < cnt) {
      tok = lists[e * NTOK + row0 + tid];
      w   = wts[e * NTOK + row0 + tid];
    }
    s_tok[tid] = tok;
    s_w[tid]   = w;
  }
  __syncthreads();

  // ---- stage gathered activations as f16 into s_mid[32][1024] -------------
  {
    const int r  = tid >> 3;                  // row 0..31 (8 threads per row)
    const int c0 = tid & 7;
    const float* xr = x + (size_t)s_tok[r] * DMODEL;
    unsigned* dst = (unsigned*)&s_mid[r][0];
    #pragma unroll
    for (int it = 0; it < 32; ++it) {
      const int c4 = c0 + it * 8;             // float4 index 0..255
      const float4 g = *(const float4*)(xr + c4 * 4);
      union { _Float16 h[2]; unsigned u; } p0, p1;
      p0.h[0] = (_Float16)g.x; p0.h[1] = (_Float16)g.y;
      p1.h[0] = (_Float16)g.z; p1.h[1] = (_Float16)g.w;
      dst[c4 * 2]     = p0.u;
      dst[c4 * 2 + 1] = p1.u;
    }
  }
  __syncthreads();                            // s_mid (xف16) stable for GEMM1

  const int wave = tid >> 5, lane = tid & 31;
  const int half = lane >> 4, lm = lane & 15; // A row / B,C col within tile
  const int cg = wave & 3,  rg = wave >> 2;
  const int colbase = cg * 256;
  const int rowbase = rg * 16;

  const float* W1e = W1 + (size_t)e * DMODEL * DFF;
  const float* W2e = W2 + (size_t)e * DFF * DMODEL;
  const _Float16* bcol = &s_B[colbase + lm][half * 16];  // tile t at +t*16*BPAD

  v8f zero;
  #pragma unroll
  for (int i = 0; i < 8; ++i) zero[i] = 0.f;

  // ---------------- GEMM1: mid = x @ W1[e], K = DMODEL ----------------------
  v8f acc[16];
  #pragma unroll
  for (int t = 0; t < 16; ++t) acc[t] = zero;

  for (int kk = 0; kk < DMODEL; kk += 32) {
    // A fragment: s_mid is read-only here; issue before the barriers so the
    // ds loads complete under the barrier/staging latency.
    const v16h af = ldfragA(&s_mid[rowbase + lm][kk + half * 16]);
    __syncthreads();                          // previous panel fully consumed
    stage_B(W1e + (size_t)kk * DFF, s_B, tid);
    if (kk + 32 < DMODEL)
      __builtin_prefetch(W1e + (size_t)(kk + 32) * DFF + tid * 16, 0, 0);
    __syncthreads();                          // panel visible

    // software pipeline: B fragment loads run 2 tiles ahead of the WMMAs
    v16h bcur = ldfragB(bcol);
    v16h bnxt = ldfragB(bcol + 16 * BPAD);
    #pragma unroll
    for (int t = 0; t < 16; ++t) {
      const v16h bpre = (t + 2 < 16) ? ldfragB(bcol + (t + 2) * 16 * BPAD) : bcur;
      acc[t] = __builtin_amdgcn_wmma_f32_16x16x32_f16(
          false, af, false, bcur, (short)0, acc[t], false, false);
      bcur = bnxt;
      bnxt = bpre;
    }
  }
  __syncthreads();                            // all A reads done before overwrite

  // squared-ReLU -> f16, overwrite s_mid with mid tile
  #pragma unroll
  for (int t = 0; t < 16; ++t) {
    #pragma unroll
    for (int r = 0; r < 8; ++r) {             // C layout: row = r + 8*half, col = lm
      float v = acc[t][r];
      v = (v > 0.f) ? v * v : 0.f;
      s_mid[rowbase + r + 8 * half][colbase + t * 16 + lm] = (_Float16)v;
    }
  }
  __syncthreads();                            // s_mid (mid f16) stable for GEMM2

  // ---------------- GEMM2: y = mid @ W2[e], K = DFF -------------------------
  v8f acc2[16];
  #pragma unroll
  for (int t = 0; t < 16; ++t) acc2[t] = zero;

  for (int kk = 0; kk < DFF; kk += 32) {
    const v16h af = ldfragA(&s_mid[rowbase + lm][kk + half * 16]);
    __syncthreads();
    stage_B(W2e + (size_t)kk * DMODEL, s_B, tid);
    if (kk + 32 < DFF)
      __builtin_prefetch(W2e + (size_t)(kk + 32) * DMODEL + tid * 16, 0, 0);
    __syncthreads();

    v16h bcur = ldfragB(bcol);
    v16h bnxt = ldfragB(bcol + 16 * BPAD);
    #pragma unroll
    for (int t = 0; t < 16; ++t) {
      const v16h bpre = (t + 2 < 16) ? ldfragB(bcol + (t + 2) * 16 * BPAD) : bcur;
      acc2[t] = __builtin_amdgcn_wmma_f32_16x16x32_f16(
          false, af, false, bcur, (short)0, acc2[t], false, false);
      bcur = bnxt;
      bnxt = bpre;
    }
  }

  // weighted combine into out (two experts per token -> atomic f32 add)
  #pragma unroll
  for (int r = 0; r < 8; ++r) {
    const int   row = rowbase + r + 8 * half;
    const float w   = s_w[row];
    if (w != 0.f) {
      float* op = out + (size_t)s_tok[row] * DMODEL + colbase + lm;
      #pragma unroll
      for (int t = 0; t < 16; ++t) atomicAdd(op + t * 16, w * acc2[t][r]);
    }
  }
}

// ---------------------------------------------------------------------------
extern "C" void kernel_launch(void* const* d_in, const int* in_sizes, int n_in,
                              void* d_out, int out_size, void* d_ws, size_t ws_size,
                              hipStream_t stream) {
  const float* x  = (const float*)d_in[0];   // [N, 1024]
  const float* Wr = (const float*)d_in[1];   // [1024, 8]
  const float* W1 = (const float*)d_in[2];   // [8, 1024, 1024]
  const float* W2 = (const float*)d_in[3];   // [8, 1024, 1024]
  float* out = (float*)d_out;                // [N, 1024]

  // workspace layout: counts (pad 256B) | lists [E*N] i32 | wts [E*N] f32
  int*   counts = (int*)d_ws;
  int*   lists  = (int*)((char*)d_ws + 256);
  float* wts    = (float*)((char*)d_ws + 256 + (size_t)NEXP * NTOK * sizeof(int));

  moe_zero_kernel<<<4096, 256, 0, stream>>>(out, counts);
  moe_router_kernel<<<NTOK / 8, 256, 0, stream>>>(x, Wr, counts, lists, wts);
  dim3 grid(NBLK, NEXP);
  moe_expert_kernel<<<grid, 256, 0, stream>>>(x, W1, W2, counts, lists, wts, out);
}
